// WindowedMLP_2516850835672
// MI455X (gfx1250) — matrix operand order repaired and checked by reading, not compiled, not run on hardware
//
#include <hip/hip_runtime.h>

typedef __bf16 bf16;
typedef __attribute__((ext_vector_type(16))) __bf16 v16bf;
typedef __attribute__((ext_vector_type(8)))  __bf16 v8bf;
typedef __attribute__((ext_vector_type(8)))  float  v8f;
typedef __attribute__((ext_vector_type(4)))  unsigned int v4u;

constexpr int kTotal = 50000;
constexpr int kWS    = 256;    // window size / hidden dim
constexpr int kNC    = 26;
constexpr int kNSC   = 7;
constexpr int kM     = 195;    // full windows
constexpr int kLast  = 80;     // ragged window width
constexpr int kInF   = 289;    // 256 + 33
constexpr int kKP1   = 320;    // 289 padded to multiple of 32
constexpr int kInL   = 113;    // 80 + 33
constexpr int kB     = 512;
constexpr int kBT    = 64;     // batch tile per block

// padded LDS strides (in bf16 elements, keep 16B alignment, dodge worst banks)
constexpr int kXST = 328;      // >= 320
constexpr int kHST = 264;      // >= 256

// ---------------------------------------------------------------------------
// f32 [M][K][256] -> bf16 [M][256][Kpad] (transpose + pad + downconvert)
// ---------------------------------------------------------------------------
__global__ void transpose_to_bf16(const float* __restrict__ W,
                                  bf16* __restrict__ WT, int K, int Kpad) {
  __shared__ float t[32][33];
  const int m  = blockIdx.x;
  const int k0 = blockIdx.y * 32;
  const int n0 = blockIdx.z * 32;
  const int tx = threadIdx.x & 31;
  const int ty = threadIdx.x >> 5;             // 0..7
  const float* Wm = W + (size_t)m * K * kWS;
  bf16* WTm = WT + (size_t)m * kWS * Kpad;
  for (int kk = ty; kk < 32; kk += 8) {
    const int k = k0 + kk;
    t[kk][tx] = (k < K) ? Wm[(size_t)k * kWS + n0 + tx] : 0.0f;
  }
  __syncthreads();
  for (int nn = ty; nn < 32; nn += 8)
    WTm[(size_t)(n0 + nn) * Kpad + k0 + tx] = (bf16)t[tx][nn];
}

// ---------------------------------------------------------------------------
// one fused GEMM layer: A (LDS, bf16 row-major) x WT (global, n-major bf16)
// K-slices of 32 staged in double-buffered LDS. Per wave: 8 wmma tiles.
// A fragment pipelined across K-steps; B fragment pipelined across tiles.
// ---------------------------------------------------------------------------
template <int NK>
__device__ __forceinline__ void gemm_layer(const bf16* __restrict__ A_lds,
                                           int a_stride,
                                           const bf16* __restrict__ WT,
                                           bf16 (*__restrict__ wbuf)[kWS * 32],
                                           int r, int c, int ln, int kh,
                                           int tid, v8f acc[8]) {
  constexpr int Kpad = NK * 32;
  // prologue: stage K-slice 0 (each thread: 32 contiguous bytes of one n-row)
  {
    const v4u* src = (const v4u*)(WT + (size_t)tid * Kpad);
    v4u a = src[0], b = src[1];
    *(v4u*)(&wbuf[0][tid * 32])      = a;
    *(v4u*)(&wbuf[0][tid * 32 + 16]) = b;
  }
  const bf16* arow = A_lds + (size_t)(r * 16 + ln) * a_stride;
  // prefetch A fragment for K-step 0 (A region is stable for the whole layer)
  v8bf alo = *(const v8bf*)(arow + kh * 16);
  v8bf ahi = *(const v8bf*)(arow + kh * 16 + 8);
  __syncthreads();

  const int nb0 = (c * 128 + ln) * 32 + kh * 16;   // B frag base, tile stride 512
  for (int kk = 0; kk < NK; ++kk) {
    v4u na, nb;
    const bool more = (kk + 1 < NK);
    if (more) {  // issue next slice's global loads before the math
      const v4u* src = (const v4u*)(WT + (size_t)tid * Kpad + (kk + 1) * 32);
      na = src[0];
      nb = src[1];
    }
    v16bf afrag = __builtin_shufflevector(alo, ahi, 0, 1, 2, 3, 4, 5, 6, 7, 8,
                                          9, 10, 11, 12, 13, 14, 15);
    const bf16* wb = wbuf[kk & 1];
    // pipelined B fragments: load b[t+1] before wmma on b[t]
    v16bf bcur = *(const v16bf*)(wb + nb0);
#pragma unroll
    for (int t = 0; t < 8; ++t) {
      v16bf bnext = bcur;
      if (t < 7) bnext = *(const v16bf*)(wb + nb0 + (t + 1) * 512);
      acc[t] = __builtin_amdgcn_wmma_f32_16x16x32_bf16(
          false, afrag, false, bcur, (short)0, acc[t], false, false);
      bcur = bnext;
    }
    if (more) {
      // prefetch next A fragment (stable LDS region, safe before barrier)
      const int K0n = (kk + 1) * 32;
      alo = *(const v8bf*)(arow + K0n + kh * 16);
      ahi = *(const v8bf*)(arow + K0n + kh * 16 + 8);
      bf16* d = &wbuf[(kk + 1) & 1][tid * 32];
      *(v4u*)(d)      = na;
      *(v4u*)(d + 16) = nb;
    }
    __syncthreads();
  }
}

__device__ __forceinline__ void store_act_lds(const v8f acc[8],
                                              const float* __restrict__ bias,
                                              bf16* __restrict__ H, int hst,
                                              int r, int c, int ln, int kh) {
#pragma unroll
  for (int t = 0; t < 8; ++t) {
    const int n = c * 128 + t * 16 + ln;
    const float bv = bias[n];
#pragma unroll
    for (int v = 0; v < 8; ++v) {
      float x = acc[t][v] + bv;
      x = (x < 0.0f) ? 0.01f * x : x;                 // leaky relu
      const int row = r * 16 + v + kh * 8;            // C/D VGPR layout
      H[row * hst + n] = (bf16)x;
    }
  }
}

// ---------------------------------------------------------------------------
// fused 3-layer window MLP, one (window, 64-row batch tile) per block
// ---------------------------------------------------------------------------
__global__ __launch_bounds__(256)
void win_mlp(const float* __restrict__ geno, const int* __restrict__ labels,
             const int* __restrict__ slabels, const bf16* __restrict__ W1T,
             const float* __restrict__ b1, const bf16* __restrict__ W2T,
             const float* __restrict__ b2, const bf16* __restrict__ W3T,
             const float* __restrict__ b3, float* __restrict__ out) {
  __shared__ bf16 xbuf[kBT * kXST];          // layer-1 input / layer-2 output
  __shared__ bf16 hbuf[kBT * kHST];          // layer-1 output
  __shared__ bf16 wbuf[2][kWS * 32];         // double-buffered weight slice

  const int m     = blockIdx.y;
  const int bbase = blockIdx.x * kBT;
  const int tid   = threadIdx.x;
  const int lane  = tid & 31;
  const int w     = tid >> 5;
  const int r     = w & 3;                   // 16-row strip within tile
  const int c     = w >> 2;                  // 128-col half
  const int ln    = lane & 15;
  const int kh    = lane >> 4;

  // ---- assemble X tile: [64 rows x 320] = genotype slice | one-hots | 0-pad
  for (int i = tid; i < kBT * 64; i += 256) {
    const int row = i >> 6, q = i & 63;
    const float4 g = *(const float4*)(geno + (size_t)(bbase + row) * kTotal +
                                      (size_t)m * kWS + q * 4);
    bf16* dst = xbuf + row * kXST + q * 4;
    dst[0] = (bf16)g.x; dst[1] = (bf16)g.y;
    dst[2] = (bf16)g.z; dst[3] = (bf16)g.w;
  }
  for (int i = tid; i < kBT * 64; i += 256) {
    const int row = i >> 6, cc = i & 63;
    const int lab = labels[bbase + row];
    const int sl  = slabels[bbase + row];
    float v = 0.0f;
    if (cc < kNC)             v = (cc == lab) ? 1.0f : 0.0f;
    else if (cc < kNC + kNSC) v = ((cc - kNC) == sl) ? 1.0f : 0.0f;
    xbuf[row * kXST + kWS + cc] = (bf16)v;
  }
  // (visibility ensured by the barrier inside gemm_layer's prologue)

  v8f zero = {};
  v8f acc[8];

  // ---- layer 1: [64x320] x [320x256]
#pragma unroll
  for (int t = 0; t < 8; ++t) acc[t] = zero;
  gemm_layer<kKP1 / 32>(xbuf, kXST, W1T + (size_t)m * kWS * kKP1, wbuf, r, c,
                        ln, kh, tid, acc);
  store_act_lds(acc, b1 + (size_t)m * kWS, hbuf, kHST, r, c, ln, kh);
  __syncthreads();

  // ---- layer 2: [64x256] x [256x256]  (write h2 into xbuf region)
#pragma unroll
  for (int t = 0; t < 8; ++t) acc[t] = zero;
  gemm_layer<kWS / 32>(hbuf, kHST, W2T + (size_t)m * kWS * kWS, wbuf, r, c, ln,
                       kh, tid, acc);
  store_act_lds(acc, b2 + (size_t)m * kWS, xbuf, kHST, r, c, ln, kh);
  __syncthreads();

  // ---- layer 3: [64x256] x [256x256] -> f32 out (bias, no relu)
#pragma unroll
  for (int t = 0; t < 8; ++t) acc[t] = zero;
  gemm_layer<kWS / 32>(xbuf, kHST, W3T + (size_t)m * kWS * kWS, wbuf, r, c, ln,
                       kh, tid, acc);
  {
    float* outp = out + (size_t)bbase * kTotal + (size_t)m * kWS;
    const float* b3m = b3 + (size_t)m * kWS;
#pragma unroll
    for (int t = 0; t < 8; ++t) {
      const int n = c * 128 + t * 16 + ln;
      const float bv = b3m[n];
#pragma unroll
      for (int v = 0; v < 8; ++v) {
        const int row = r * 16 + v + kh * 8;
        outp[(size_t)row * kTotal + n] = acc[t][v] + bv;
      }
    }
  }
}

// ---------------------------------------------------------------------------
// last ragged 80-wide window: negligible FLOPs, plain f32, one row per block
// ---------------------------------------------------------------------------
__global__ __launch_bounds__(128)
void last_mlp(const float* __restrict__ geno, const int* __restrict__ labels,
              const int* __restrict__ slabels, const float* __restrict__ lW1,
              const float* __restrict__ lb1, const float* __restrict__ lW2,
              const float* __restrict__ lb2, const float* __restrict__ lW3,
              const float* __restrict__ lb3, float* __restrict__ out) {
  __shared__ float x[128];
  __shared__ float h[96];
  const int b = blockIdx.x;
  const int t = threadIdx.x;
  if (t < kInL) {
    float v;
    if (t < kLast)            v = geno[(size_t)b * kTotal + kM * kWS + t];
    else if (t < kLast + kNC) v = (labels[b] == (t - kLast)) ? 1.0f : 0.0f;
    else                      v = (slabels[b] == (t - kLast - kNC)) ? 1.0f : 0.0f;
    x[t] = v;
  }
  __syncthreads();
  if (t < kLast) {
    float s = lb1[t];
    for (int k = 0; k < kInL; ++k) s += x[k] * lW1[k * kLast + t];
    h[t] = (s < 0.0f) ? 0.01f * s : s;
  }
  __syncthreads();
  if (t < kLast) {
    float s = lb2[t];
    for (int k = 0; k < kLast; ++k) s += h[k] * lW2[k * kLast + t];
    x[t] = (s < 0.0f) ? 0.01f * s : s;
  }
  __syncthreads();
  if (t < kLast) {
    float s = lb3[t];
    for (int k = 0; k < kLast; ++k) s += x[k] * lW3[k * kLast + t];
    out[(size_t)b * kTotal + kM * kWS + t] = s;
  }
}

// ---------------------------------------------------------------------------
extern "C" void kernel_launch(void* const* d_in, const int* in_sizes, int n_in,
                              void* d_out, int out_size, void* d_ws,
                              size_t ws_size, hipStream_t stream) {
  const float* geno    = (const float*)d_in[0];
  const int*   labels  = (const int*)d_in[1];
  const int*   slabels = (const int*)d_in[2];
  const float* W1 = (const float*)d_in[3];
  const float* b1 = (const float*)d_in[4];
  const float* W2 = (const float*)d_in[5];
  const float* b2 = (const float*)d_in[6];
  const float* W3 = (const float*)d_in[7];
  const float* b3 = (const float*)d_in[8];
  const float* lW1 = (const float*)d_in[9];
  const float* lb1 = (const float*)d_in[10];
  const float* lW2 = (const float*)d_in[11];
  const float* lb2 = (const float*)d_in[12];
  const float* lW3 = (const float*)d_in[13];
  const float* lb3 = (const float*)d_in[14];
  float* out = (float*)d_out;

  // workspace: bf16 n-major weights (written fresh every call, ~83 MB)
  bf16* W1T = (bf16*)d_ws;                             // [195][256][320]
  bf16* W2T = W1T + (size_t)kM * kWS * kKP1;           // [195][256][256]
  bf16* W3T = W2T + (size_t)kM * kWS * kWS;            // [195][256][256]

  transpose_to_bf16<<<dim3(kM, kKP1 / 32, kWS / 32), 256, 0, stream>>>(
      W1, W1T, kInF, kKP1);
  transpose_to_bf16<<<dim3(kM, kWS / 32, kWS / 32), 256, 0, stream>>>(
      W2, W2T, kWS, kWS);
  transpose_to_bf16<<<dim3(kM, kWS / 32, kWS / 32), 256, 0, stream>>>(
      W3, W3T, kWS, kWS);

  win_mlp<<<dim3(kB / kBT, kM), 256, 0, stream>>>(
      geno, labels, slabels, W1T, b1, W2T, b2, W3T, b3, out);

  last_mlp<<<dim3(kB), 128, 0, stream>>>(geno, labels, slabels, lW1, lb1, lW2,
                                         lb2, lW3, lb3, out);
}
